// NCA_55250459296235
// MI455X (gfx1250) — matrix operand order repaired and checked by reading, not compiled, not run on hardware
//
#include <hip/hip_runtime.h>
#include <math.h>

// Sizes from the reference
#define N_PTS   50000
#define KNN_K   16
#define HALFD   64
#define DIM     128
// fused kernel geometry
#define WAVES_PER_BLK 8
#define FUSED_BLOCKS  3125            // 25000 waves -> exactly 2 tiles each
#define ROW_US        136             // 128 cols + 8 halfword pad (bank-conflict avoidance)
#define TILE_US       (16 * ROW_US)   // one 16x128 bf16 tile per wave

#define AS1 __attribute__((address_space(1)))   // global

typedef __bf16        v16bf __attribute__((ext_vector_type(16)));
typedef float         v8f   __attribute__((ext_vector_type(8)));
typedef unsigned int  v4u   __attribute__((ext_vector_type(4)));

struct U8 { v4u lo, hi; };

__device__ __forceinline__ unsigned short f2bf(float f) {
    unsigned int x = __float_as_uint(f);
    x += 0x7fffu + ((x >> 16) & 1u);          // round-to-nearest-even
    return (unsigned short)(x >> 16);
}
__device__ __forceinline__ unsigned int pack2(float a, float b) {
    return (unsigned int)f2bf(a) | ((unsigned int)f2bf(b) << 16);
}
__device__ __forceinline__ v16bf mkfrag(v4u lo, v4u hi) {
    U8 t; t.lo = lo; t.hi = hi;
    return __builtin_bit_cast(v16bf, t);
}

// ---------------------------------------------------------------------------
// Kernel 0: swizzle W3a/W3b (f32 row-major [k][n], 128x128) into bf16 fragment
// blobs so a wave can fetch a lane's whole 16-element B fragment as 32
// contiguous bytes.  B layout (v_wmma_f32_16x16x32_bf16, wave32):
//   lane 0-15 : n = lane,    K = 0..7, 16..23
//   lane 16-31: n = lane-16, K = 8..15, 24..31
// blob index = ((nt*4 + kb)*32 + lane)*16 + e,
// k = kb*32 + (lane>>4)*8 + (e&7) + (e>>3)*16,
// n = nt*16 + (lane&15).
// ---------------------------------------------------------------------------
__global__ void nca_swizzle_w(const float* __restrict__ W3a,
                              const float* __restrict__ W3b,
                              unsigned short* __restrict__ blob) {
    int id = blockIdx.x * blockDim.x + threadIdx.x;
    if (id >= 2 * 16384) return;
    int mat  = id >> 14;
    int rem  = id & 16383;
    int e    = rem & 15;
    int lane = (rem >> 4) & 31;
    int kb   = (rem >> 9) & 3;
    int nt   = rem >> 11;
    int n = nt * 16 + (lane & 15);
    int k = kb * 32 + (lane >> 4) * 8 + (e & 7) + (e >> 3) * 16;
    const float* W = mat ? W3b : W3a;
    blob[id] = f2bf(W[k * DIM + n]);
}

// ---------------------------------------------------------------------------
// Kernel 1: p0 = xyz @ W1 + b1     (N x 64, f32)
// ---------------------------------------------------------------------------
__global__ void nca_p0(const float* __restrict__ xyz,
                       const float* __restrict__ W1,
                       const float* __restrict__ b1,
                       float* __restrict__ p0) {
    int id = blockIdx.x * blockDim.x + threadIdx.x;
    if (id >= N_PTS * HALFD) return;
    int i = id >> 6, j = id & 63;
    float x0 = xyz[i * 3 + 0], x1 = xyz[i * 3 + 1], x2 = xyz[i * 3 + 2];
    p0[id] = b1[j] + x0 * W1[j] + x1 * W1[64 + j] + x2 * W1[128 + j];
}

// ---------------------------------------------------------------------------
// Kernel 2: p1 = p0 @ W2 + b2 ;  p_local = max_k p0[knn[i,k]] - p0[i]
// ---------------------------------------------------------------------------
__global__ void nca_p1_plocal(const float* __restrict__ p0,
                              const float* __restrict__ W2,
                              const float* __restrict__ b2,
                              const int* __restrict__ knn,
                              float* __restrict__ p1,
                              float* __restrict__ ploc) {
    int id = blockIdx.x * blockDim.x + threadIdx.x;
    if (id >= N_PTS * HALFD) return;
    int i = id >> 6, j = id & 63;
    const float* pr = p0 + (size_t)i * HALFD;
    float acc = b2[j];
#pragma unroll
    for (int c = 0; c < HALFD; ++c) acc += pr[c] * W2[c * HALFD + j];
    p1[id] = acc;
    float m = -3.402823466e+38f;
#pragma unroll
    for (int k = 0; k < KNN_K; ++k) {
        int nb = knn[i * KNN_K + k];
        m = fmaxf(m, p0[(size_t)nb * HALFD + j]);
    }
    ploc[id] = m - pr[j];
}

// ---------------------------------------------------------------------------
// Fused kernel: per point-tile (16 neighbor rows x 128 feat):
//   build bf16 feat tile in LDS -> GEMM1 (WMMA bf16, f32 acc) -> +b3a ->
//   exact GELU -> bf16 h back into same LDS tile -> GEMM2 -> +b3b -> global.
// A-fragment layout (16-bit A 16x32): lane m=lane&15, K split at 8 by lane
// half; fragment = two 16B LDS loads at k-bytes (kb*64 + hilo*16) and +32.
//
// Weight-blob pointer is laundered through `asm volatile` each iteration
// (defeats LICM that previously hoisted 512 VGPRs of fragments / spilled to
// scratch), then cast back to an address_space(1) pointer so the loads lower
// to global_load_b128 (LOADcnt only) instead of flat_load_b128 (which also
// ties up DScnt and the LDS path).
// ---------------------------------------------------------------------------
__global__ __launch_bounds__(WAVES_PER_BLK * 32, 4)
void nca_fused(const float* __restrict__ p1,
               const float* __restrict__ ploc,
               const int* __restrict__ knn,
               const unsigned short* __restrict__ wblob,
               const float* __restrict__ b3a,
               const float* __restrict__ b3b,
               float* __restrict__ out) {
    __shared__ __align__(16) unsigned short lds[WAVES_PER_BLK * TILE_US];
    const int lane = threadIdx.x & 31;
    const int wave = threadIdx.x >> 5;
    unsigned short* myT = lds + wave * TILE_US;
    const int m16  = lane & 15;
    const int hilo = lane >> 4;
    const int gw   = blockIdx.x * WAVES_PER_BLK + wave;
    const int nwaves = FUSED_BLOCKS * WAVES_PER_BLK;

    for (int tile = gw; tile < N_PTS; tile += nwaves) {
        // Launder the weight pointer: redefine it every iteration so the
        // fragment loads below cannot be hoisted out of the tile loop.
        unsigned long wbL = (unsigned long)wblob;
        asm volatile("" : "+s"(wbL));
        const unsigned long wbL2 = wbL + 32768;   // W3b half (bytes)

        // ---- build feat tile: rows t = neighbor k of point `tile` ----
        {
            const int t = lane >> 1, h = lane & 1;        // 2 lanes per row
            const int nbr = knn[tile * KNN_K + t];
            const float* prC = p1 + (size_t)tile * HALFD + h * 32;
            const float* prN = p1 + (size_t)nbr  * HALFD + h * 32;
            unsigned short* rowp = myT + t * ROW_US + h * 32;
#pragma unroll
            for (int j = 0; j < 8; ++j) {
                float4 a = *(const float4*)(prN + j * 4);
                float4 c = *(const float4*)(prC + j * 4);
                uint2 pk;
                pk.x = pack2(a.x - c.x, a.y - c.y);
                pk.y = pack2(a.z - c.z, a.w - c.w);
                *(uint2*)(rowp + j * 4) = pk;
            }
            // right half: broadcast p_local row into all 16 tile rows
            float2 pl = *(const float2*)(ploc + (size_t)tile * HALFD + lane * 2);
            unsigned int pk = pack2(pl.x, pl.y);
            unsigned short* cp = myT + HALFD + lane * 2;
#pragma unroll
            for (int r = 0; r < 16; ++r) *(unsigned int*)(cp + r * ROW_US) = pk;
        }

        // ---- load A fragments of feat (4 K-blocks) into registers ----
        U8 af[4];
        {
            const char* base = (const char*)myT + m16 * (ROW_US * 2);
#pragma unroll
            for (int kb = 0; kb < 4; ++kb) {
                const char* p = base + kb * 64 + hilo * 16;
                af[kb].lo = *(const v4u*)p;
                af[kb].hi = *(const v4u*)(p + 32);
            }
        }

        // ---- layer 1: h = gelu(feat @ W3a + b3a), bf16 -> same LDS tile ----
#pragma unroll
        for (int nt = 0; nt < 8; ++nt) {
            // two independent 2-deep WMMA chains (halve XDL RAW depth)
            v8f acc0 = {0.f, 0.f, 0.f, 0.f, 0.f, 0.f, 0.f, 0.f};
            v8f acc1 = {0.f, 0.f, 0.f, 0.f, 0.f, 0.f, 0.f, 0.f};
#pragma unroll
            for (int kk = 0; kk < 2; ++kk) {
                const int kbA = kk, kbB = kk + 2;
                const AS1 v4u* wpA = (const AS1 v4u*)
                    (wbL + (unsigned long)(((nt * 4 + kbA) * 32 + lane) << 5));
                const AS1 v4u* wpB = (const AS1 v4u*)
                    (wbL + (unsigned long)(((nt * 4 + kbB) * 32 + lane) << 5));
                acc0 = __builtin_amdgcn_wmma_f32_16x16x32_bf16(
                    false, mkfrag(af[kbA].lo, af[kbA].hi), false,
                    mkfrag(wpA[0], wpA[1]), (short)0, acc0, false, false);
                acc1 = __builtin_amdgcn_wmma_f32_16x16x32_bf16(
                    false, mkfrag(af[kbB].lo, af[kbB].hi), false,
                    mkfrag(wpB[0], wpB[1]), (short)0, acc1, false, false);
            }
            const int ncol = nt * 16 + m16;
            const float bias = b3a[ncol];
#pragma unroll
            for (int r = 0; r < 8; ++r) {
                float x = acc0[r] + acc1[r] + bias;
                float g = 0.5f * x * (1.0f + erff(x * 0.7071067811865475f));
                myT[(r + 8 * hilo) * ROW_US + ncol] = f2bf(g);
            }
        }

        // ---- layer 2: out = h @ W3b + b3b ----
        {
            const char* base = (const char*)myT + m16 * (ROW_US * 2);
#pragma unroll
            for (int kb = 0; kb < 4; ++kb) {
                const char* p = base + kb * 64 + hilo * 16;
                af[kb].lo = *(const v4u*)p;
                af[kb].hi = *(const v4u*)(p + 32);
            }
        }
        float* orow = out + (size_t)tile * (KNN_K * DIM);
#pragma unroll
        for (int nt = 0; nt < 8; ++nt) {
            v8f acc0 = {0.f, 0.f, 0.f, 0.f, 0.f, 0.f, 0.f, 0.f};
            v8f acc1 = {0.f, 0.f, 0.f, 0.f, 0.f, 0.f, 0.f, 0.f};
#pragma unroll
            for (int kk = 0; kk < 2; ++kk) {
                const int kbA = kk, kbB = kk + 2;
                const AS1 v4u* wpA = (const AS1 v4u*)
                    (wbL2 + (unsigned long)(((nt * 4 + kbA) * 32 + lane) << 5));
                const AS1 v4u* wpB = (const AS1 v4u*)
                    (wbL2 + (unsigned long)(((nt * 4 + kbB) * 32 + lane) << 5));
                acc0 = __builtin_amdgcn_wmma_f32_16x16x32_bf16(
                    false, mkfrag(af[kbA].lo, af[kbA].hi), false,
                    mkfrag(wpA[0], wpA[1]), (short)0, acc0, false, false);
                acc1 = __builtin_amdgcn_wmma_f32_16x16x32_bf16(
                    false, mkfrag(af[kbB].lo, af[kbB].hi), false,
                    mkfrag(wpB[0], wpB[1]), (short)0, acc1, false, false);
            }
            const int ncol = nt * 16 + m16;
            const float bias = b3b[ncol];
            float ov[8];
#pragma unroll
            for (int r = 0; r < 8; ++r) ov[r] = acc0[r] + acc1[r] + bias;
#pragma unroll
            for (int r = 0; r < 8; ++r)
                orow[(r + 8 * hilo) * DIM + ncol] = ov[r];
        }
    }
}

// ---------------------------------------------------------------------------
extern "C" void kernel_launch(void* const* d_in, const int* in_sizes, int n_in,
                              void* d_out, int out_size, void* d_ws, size_t ws_size,
                              hipStream_t stream) {
    (void)in_sizes; (void)n_in; (void)out_size; (void)ws_size;
    const float* xyz = (const float*)d_in[0];
    const int*   knn = (const int*)d_in[1];
    const float* W1  = (const float*)d_in[2];
    const float* b1  = (const float*)d_in[3];
    const float* W2  = (const float*)d_in[4];
    const float* b2  = (const float*)d_in[5];
    const float* W3a = (const float*)d_in[6];
    const float* b3a = (const float*)d_in[7];
    const float* W3b = (const float*)d_in[8];
    const float* b3b = (const float*)d_in[9];
    float* out = (float*)d_out;

    char* ws = (char*)d_ws;
    float* p0   = (float*)(ws);                       // 12.8 MB
    float* p1   = (float*)(ws + 12800000);            // 12.8 MB
    float* ploc = (float*)(ws + 25600000);            // 12.8 MB
    unsigned short* wblob = (unsigned short*)(ws + 38400000); // 64 KB bf16

    nca_swizzle_w<<<128, 256, 0, stream>>>(W3a, W3b, wblob);
    nca_p0<<<(N_PTS * HALFD + 255) / 256, 256, 0, stream>>>(xyz, W1, b1, p0);
    nca_p1_plocal<<<(N_PTS * HALFD + 255) / 256, 256, 0, stream>>>(
        p0, W2, b2, knn, p1, ploc);
    nca_fused<<<FUSED_BLOCKS, WAVES_PER_BLK * 32, 0, stream>>>(
        p1, ploc, knn, wblob, b3a, b3b, out);
}
